// attention_net_32487132627119
// MI455X (gfx1250) — matrix operand order repaired and checked by reading, not compiled, not run on hardware
//
#include <hip/hip_runtime.h>
#include <hip/hip_bf16.h>
#include <stdint.h>

typedef __bf16 bf16_t;
typedef __attribute__((ext_vector_type(16))) __bf16 v16bf;
typedef __attribute__((ext_vector_type(8)))  float  v8f;

#define NCH 128
#define NA 1614
#define NEG_INF (-3.402823466e38f)

union FragU { uint32_t u[8]; v16bf v; };

static __device__ __forceinline__ uint16_t f2bf_rne(float f) {
    union { float f; uint32_t u; } x; x.f = f;
    uint32_t r = x.u + 0x7FFFu + ((x.u >> 16) & 1u);
    return (uint16_t)(r >> 16);
}

__global__ void k_zero(uint32_t* __restrict__ p, int n) {
    int i = blockIdx.x * blockDim.x + threadIdx.x;
    if (i < n) p[i] = 0u;
}

// ---------------- NCHW f32 -> NHWC bf16 ----------------
__global__ void k_cvt_x(const float* __restrict__ in, bf16_t* __restrict__ out,
                        int C, int HW, int total) {
    int idx = blockIdx.x * blockDim.x + threadIdx.x;
    if (idx >= total) return;
    int c = idx % C;
    int p = (idx / C) % HW;
    int b = idx / (C * HW);
    out[idx] = (bf16_t)in[(b * C + c) * HW + p];
}

// ------- pack OIHW f32 weights into WMMA B-fragment order (bf16 pairs) -------
__global__ void k_pack_w(const float* __restrict__ w, uint32_t* __restrict__ out,
                         int K, int KC, int total) {
    int idx = blockIdx.x * blockDim.x + threadIdx.x;
    if (idx >= total) return;
    int j    = idx & 7;
    int lane = (idx >> 3) & 31;
    int rest = idx >> 8;
    int kc   = rest % KC; rest /= KC;
    int nt   = rest & 7;
    int tap  = rest >> 3;
    int n  = nt * 16 + (lane & 15);
    int k0 = kc * 32 + ((lane >> 4) << 4) + 2 * j;
    float a = w[((size_t)(n * K + k0)) * 9 + tap];
    float b = w[((size_t)(n * K + k0 + 1)) * 9 + tap];
    out[idx] = (uint32_t)f2bf_rne(a) | ((uint32_t)f2bf_rne(b) << 16);
}

// ------- implicit-GEMM conv (3x3, pad 1): async-to-LDS staged, WMMA bf16 -------
// block = 256 (8 waves) covers 4 M-tiles x 128 N. Per (tap,kc) step the block
// stages 4KB of A rows + 8KB of B fragments into LDS exactly once via
// global_load_async_to_lds_b128 (double buffered, ASYNCcnt pipelined), then all
// waves consume via ds_load_b128. Wave w: mtile bx*4+(w>>1), N-half w&1,
// 4 accumulators -> 4 WMMAs per step per wave.
__global__ void __launch_bounds__(256)
k_conv_wmma(const bf16_t* __restrict__ X, const uint32_t* __restrict__ Wp,
            const float* __restrict__ bias, bf16_t* __restrict__ Y,
            const uint8_t* __restrict__ zbuf,
            int IH, int IW, int OH, int OW, int K, int stride, int MT) {
    __shared__ __align__(16) uint8_t smem[2][12 * 1024];
    const int KC   = K >> 5;
    const int tid  = threadIdx.x;
    const int lane = tid & 31;
    const int wv   = tid >> 5;
    const int hw   = OH * OW;

    // ---- consumer identity ----
    int mtile = blockIdx.x * 4 + (wv >> 1);
    if (mtile >= MT) mtile = MT - 1;          // clamp: keep all waves in barriers
    const int nh  = wv & 1;
    const int hi  = lane >> 4;
    const int row = mtile * 16 + (lane & 15);
    const int b   = row / hw;
    const int p   = row % hw;
    const int oy  = p / OW, ox = p % OW;

    // ---- producer identity: one A chunk + two B chunks per thread ----
    const int am = tid >> 6, ar = (tid >> 2) & 15, aseg = tid & 3;
    int mtA = blockIdx.x * 4 + am;
    if (mtA >= MT) mtA = MT - 1;
    const int rowA = mtA * 16 + ar;
    const int bA = rowA / hw, pA = rowA % hw;
    const int oyA = pA / OW, oxA = pA % OW;
    const uint32_t ldsA  = (uint32_t)(am * 1024 + ar * 64 + aseg * 16);
    const int cid0 = tid * 2,     f0 = cid0 >> 6, seg0 = cid0 & 63;
    const int cid1 = tid * 2 + 1, f1 = cid1 >> 6, seg1 = cid1 & 63;
    const uint32_t ldsB0 = (uint32_t)(4096 + f0 * 1024 + seg0 * 16);
    const uint32_t ldsB1 = (uint32_t)(4096 + f1 * 1024 + seg1 * 16);
    const uint32_t smBase = (uint32_t)(uintptr_t)(&smem[0][0]);

    v8f acc[4];
#pragma unroll
    for (int t = 0; t < 4; ++t) acc[t] = (v8f){};

    for (int tap = 0; tap < 9; ++tap) {
        const int dy = tap / 3, dx = tap % 3;
        // producer A source (zero-page for padding rows, stride 0)
        const int iyA = oyA * stride + dy - 1;
        const int ixA = oxA * stride + dx - 1;
        const bool vA = (iyA >= 0) && (iyA < IH) && (ixA >= 0) && (ixA < IW);
        const uint8_t* srcA =
            vA ? (const uint8_t*)(X + (size_t)((bA * IH + iyA) * IW + ixA) * K) + aseg * 16
               : zbuf;
        const size_t stepA = vA ? 64 : 0;
        // producer B sources (packed fragments, 1KB per kc)
        const uint8_t* srcB0 = (const uint8_t*)(Wp + (size_t)(tap * 8 + f0) * KC * 256) + seg0 * 16;
        const uint8_t* srcB1 = (const uint8_t*)(Wp + (size_t)(tap * 8 + f1) * KC * 256) + seg1 * 16;

        // prologue: stage kc=0 into buffer 0
        asm volatile("global_load_async_to_lds_b128 %0, %1, off"
                     :: "v"(smBase + ldsA), "v"(srcA) : "memory");
        asm volatile("global_load_async_to_lds_b128 %0, %1, off"
                     :: "v"(smBase + ldsB0), "v"(srcB0) : "memory");
        asm volatile("global_load_async_to_lds_b128 %0, %1, off"
                     :: "v"(smBase + ldsB1), "v"(srcB1) : "memory");

        for (int kc = 0; kc < KC; ++kc) {
            if (kc + 1 < KC) {   // issue next step into the other buffer
                const uint32_t nb = smBase + (uint32_t)(((kc + 1) & 1) * 12288);
                asm volatile("global_load_async_to_lds_b128 %0, %1, off"
                             :: "v"(nb + ldsA), "v"(srcA + (size_t)(kc + 1) * stepA) : "memory");
                asm volatile("global_load_async_to_lds_b128 %0, %1, off"
                             :: "v"(nb + ldsB0), "v"(srcB0 + (size_t)(kc + 1) * 1024) : "memory");
                asm volatile("global_load_async_to_lds_b128 %0, %1, off"
                             :: "v"(nb + ldsB1), "v"(srcB1 + (size_t)(kc + 1) * 1024) : "memory");
                asm volatile("s_wait_asynccnt 0x3" ::: "memory");  // current buf done, next in flight
            } else {
                asm volatile("s_wait_asynccnt 0x0" ::: "memory");
            }
            __syncthreads();

            const uint8_t* buf = &smem[kc & 1][0];
            FragU af;
            const uint8_t* ap = buf + ((wv >> 1) * 1024 + (lane & 15) * 64 + hi * 32);
            *(uint4*)&af.u[0] = *(const uint4*)(ap);
            *(uint4*)&af.u[4] = *(const uint4*)(ap + 16);
#pragma unroll
            for (int t = 0; t < 4; ++t) {
                FragU bq;
                const uint8_t* bp = buf + (4096 + (nh * 4 + t) * 1024 + lane * 32);
                *(uint4*)&bq.u[0] = *(const uint4*)(bp);
                *(uint4*)&bq.u[4] = *(const uint4*)(bp + 16);
                acc[t] = __builtin_amdgcn_wmma_f32_16x16x32_bf16(
                    false, af.v, false, bq.v, (short)0, acc[t], false, false);
            }
            __syncthreads();   // all waves done reading before buffer reuse
        }
    }

    // C/D layout: lanes 0-15 rows 0..7 in v0..7, lanes 16-31 rows 8..15.
#pragma unroll
    for (int t = 0; t < 4; ++t) {
        const int n  = nh * 64 + t * 16 + (lane & 15);
        const float bv = bias[n];
#pragma unroll
        for (int j = 0; j < 8; ++j) {
            int m  = j + (hi << 3);
            int r2 = mtile * 16 + m;
            float v = acc[t][j] + bv;
            v = v > 0.f ? v : 0.f;
            Y[(size_t)r2 * NCH + n] = (bf16_t)v;
        }
    }
}

// ---------------- 1x1 conv score heads ----------------
__global__ void k_head(const bf16_t* __restrict__ D, const float* __restrict__ wt,
                       const float* __restrict__ bt, float* __restrict__ out,
                       int HW, int CO, int out_off, int total) {
    int idx = blockIdx.x * blockDim.x + threadIdx.x;
    if (idx >= total) return;
    int p  = idx % HW;
    int co = (idx / HW) % CO;
    int b  = idx / (HW * CO);
    const bf16_t* d = D + (size_t)(b * HW + p) * NCH;
    const float*  w = wt + co * NCH;
    float s = 0.f;
#pragma unroll 4
    for (int c = 0; c < NCH; ++c) s += (float)d[c] * w[c];
    out[b * NA + out_off + co * HW + p] = s + bt[co];
}

// ---------------- greedy hard-NMS, one wave32 per batch ----------------
__global__ void k_nms(const float* __restrict__ scores, const int* __restrict__ anchors,
                      float* __restrict__ out_idx, float* __restrict__ out_prob,
                      int* __restrict__ boxes_sel) {
    __shared__ float s[NA];
    __shared__ float bmax[32];
    __shared__ int   bidx[32];
    __shared__ int   selS;
    __shared__ float sb[4], areaS;
    int b = blockIdx.x;
    int lane = threadIdx.x;
    for (int i = lane; i < NA; i += 32) s[i] = scores[b * NA + i];
    __syncthreads();
    for (int t = 0; t < 6; ++t) {
        float m = NEG_INF; int mi = 0;
        for (int i = lane; i < NA; i += 32) { float v = s[i]; if (v > m) { m = v; mi = i; } }
        bmax[lane] = m; bidx[lane] = mi;
        __syncthreads();
        if (lane == 0) {
            float mm = NEG_INF; int ii = 0;
            for (int l = 0; l < 32; ++l) if (bmax[l] > mm) { mm = bmax[l]; ii = bidx[l]; }
            selS = ii;
            out_idx[b * 6 + t]  = (float)ii;
            out_prob[b * 6 + t] = mm;
            int y0 = anchors[ii * 4 + 0], x0 = anchors[ii * 4 + 1];
            int y1 = anchors[ii * 4 + 2], x1 = anchors[ii * 4 + 3];
            boxes_sel[(b * 6 + t) * 4 + 0] = y0;
            boxes_sel[(b * 6 + t) * 4 + 1] = x0;
            boxes_sel[(b * 6 + t) * 4 + 2] = y1;
            boxes_sel[(b * 6 + t) * 4 + 3] = x1;
            sb[0] = (float)y0; sb[1] = (float)x0; sb[2] = (float)y1; sb[3] = (float)x1;
            areaS = (sb[2] - sb[0]) * (sb[3] - sb[1]);
        }
        __syncthreads();
        int sel = selS;
        float by0 = sb[0], bx0 = sb[1], by1 = sb[2], bx1 = sb[3], ar = areaS;
        for (int i = lane; i < NA; i += 32) {
            float ay0 = (float)anchors[i * 4 + 0], ax0 = (float)anchors[i * 4 + 1];
            float ay1 = (float)anchors[i * 4 + 2], ax1 = (float)anchors[i * 4 + 3];
            float yy0 = fmaxf(by0, ay0), xx0 = fmaxf(bx0, ax0);
            float yy1 = fminf(by1, ay1), xx1 = fminf(bx1, ax1);
            float inter = fmaxf(yy1 - yy0, 0.f) * fmaxf(xx1 - xx0, 0.f);
            float ai = (ay1 - ay0) * (ax1 - ax0);
            float iou = inter / (ai + ar - inter);
            if (iou > 0.25f) s[i] = NEG_INF;
        }
        if (lane == 0) s[sel] = NEG_INF;
        __syncthreads();
    }
}

// ---------------- bilinear crop-resize from implicitly padded image ----------------
static __device__ __forceinline__ float samp_img(const float* __restrict__ x, int bc,
                                                 int yi, int xi) {
    int iy = yi - 224, ix = xi - 224;
    if ((unsigned)iy >= 448u || (unsigned)ix >= 448u) return 0.f;
    return x[((size_t)bc * 448 + iy) * 448 + ix];
}

__global__ void k_crop(const float* __restrict__ x, const int* __restrict__ boxes,
                       float* __restrict__ out, int total) {
    int idx = blockIdx.x * blockDim.x + threadIdx.x;
    if (idx >= total) return;
    int ox = idx % 224;
    int oy = (idx / 224) % 224;
    int c  = (idx / (224 * 224)) % 3;
    int n  = (idx / (224 * 224 * 3)) % 6;
    int b  = idx / (224 * 224 * 3 * 6);
    const int* bx = boxes + (b * 6 + n) * 4;
    float y0 = (float)bx[0], x0 = (float)bx[1], y1 = (float)bx[2], x1 = (float)bx[3];
    float h = y1 - y0, w = x1 - x0;
    float ys = y0 + ((float)oy / 223.f) * (h - 1.f);
    float xs = x0 + ((float)ox / 223.f) * (w - 1.f);
    float yf = floorf(ys), xf = floorf(xs);
    float wy = ys - yf, wx = xs - xf;
    int yi0 = min(max((int)yf, 0), 895), yi1 = min(yi0 + 1, 895);
    int xi0 = min(max((int)xf, 0), 895), xi1 = min(xi0 + 1, 895);
    int bc = b * 3 + c;
    float v00 = samp_img(x, bc, yi0, xi0), v01 = samp_img(x, bc, yi0, xi1);
    float v10 = samp_img(x, bc, yi1, xi0), v11 = samp_img(x, bc, yi1, xi1);
    float top = v00 * (1.f - wx) + v01 * wx;
    float bot = v10 * (1.f - wx) + v11 * wx;
    out[idx] = top * (1.f - wy) + bot * wy;
}

extern "C" void kernel_launch(void* const* d_in, const int* in_sizes, int n_in,
                              void* d_out, int out_size, void* d_ws, size_t ws_size,
                              hipStream_t stream) {
    const float* x    = (const float*)d_in[0];
    const float* rpn  = (const float*)d_in[1];
    const float* w_d1 = (const float*)d_in[2];
    const float* b_d1 = (const float*)d_in[3];
    const float* w_d2 = (const float*)d_in[4];
    const float* b_d2 = (const float*)d_in[5];
    const float* w_d3 = (const float*)d_in[6];
    const float* b_d3 = (const float*)d_in[7];
    const float* w_t1 = (const float*)d_in[8];
    const float* b_t1 = (const float*)d_in[9];
    const float* w_t2 = (const float*)d_in[10];
    const float* b_t2 = (const float*)d_in[11];
    const float* w_t3 = (const float*)d_in[12];
    const float* b_t3 = (const float*)d_in[13];
    const int* anchors = (const int*)d_in[14];
    float* out = (float*)d_out;

    char* ws = (char*)d_ws;
    size_t off = 0;
    auto alloc = [&](size_t bytes) -> char* {
        char* p = ws + off;
        off += (bytes + 255) & ~(size_t)255;
        return p;
    };
    bf16_t*   Xbf = (bf16_t*)alloc(16ull * 196 * 2048 * 2);
    uint32_t* W1p = (uint32_t*)alloc(9ull * 8 * 64 * 1024);
    bf16_t*   D1  = (bf16_t*)alloc(16ull * 196 * 128 * 2);
    uint32_t* W2p = (uint32_t*)alloc(9ull * 8 * 4 * 1024);
    bf16_t*   D2  = (bf16_t*)alloc(16ull * 49 * 128 * 2);
    uint32_t* W3p = (uint32_t*)alloc(9ull * 8 * 4 * 1024);
    bf16_t*   D3  = (bf16_t*)alloc(16ull * 16 * 128 * 2);
    int*  boxesSel = (int*)alloc(16 * 6 * 4 * sizeof(int));
    uint8_t* zbuf  = (uint8_t*)alloc(256);

    k_zero<<<1, 64, 0, stream>>>((uint32_t*)zbuf, 64);
    { int tot = 16 * 196 * 2048;
      k_cvt_x<<<(tot + 255) / 256, 256, 0, stream>>>(rpn, Xbf, 2048, 196, tot); }
    { int tot = 9 * 8 * 64 * 256;
      k_pack_w<<<(tot + 255) / 256, 256, 0, stream>>>(w_d1, W1p, 2048, 64, tot); }
    k_conv_wmma<<<49, 256, 0, stream>>>(Xbf, W1p, b_d1, D1, zbuf, 14, 14, 14, 14, 2048, 1, 196);
    { int tot = 9 * 8 * 4 * 256;
      k_pack_w<<<(tot + 255) / 256, 256, 0, stream>>>(w_d2, W2p, 128, 4, tot); }
    k_conv_wmma<<<13, 256, 0, stream>>>(D1, W2p, b_d2, D2, zbuf, 14, 14, 7, 7, 128, 2, 49);
    { int tot = 9 * 8 * 4 * 256;
      k_pack_w<<<(tot + 255) / 256, 256, 0, stream>>>(w_d3, W3p, 128, 4, tot); }
    k_conv_wmma<<<4, 256, 0, stream>>>(D2, W3p, b_d3, D3, zbuf, 7, 7, 4, 4, 128, 2, 16);

    { int tot = 16 * 6 * 196;
      k_head<<<(tot + 255) / 256, 256, 0, stream>>>(D1, w_t1, b_t1, out, 196, 6, 0, tot); }
    { int tot = 16 * 6 * 49;
      k_head<<<(tot + 255) / 256, 256, 0, stream>>>(D2, w_t2, b_t2, out, 49, 6, 1176, tot); }
    { int tot = 16 * 9 * 16;
      k_head<<<(tot + 255) / 256, 256, 0, stream>>>(D3, w_t3, b_t3, out, 16, 9, 1470, tot); }

    k_nms<<<16, 32, 0, stream>>>(out, anchors, out + 16 * NA, out + 16 * NA + 96, boxesSel);

    { int tot = 16 * 6 * 3 * 224 * 224;
      k_crop<<<(tot + 255) / 256, 256, 0, stream>>>(x, boxesSel, out + 26016, tot); }
}
